// RandomCausalMHA_xformers_9646496547175
// MI455X (gfx1250) — matrix operand order repaired and checked by reading, not compiled, not run on hardware
//
#include <hip/hip_runtime.h>

// Problem constants (from reference)
#define BB 4
#define TT 2048
#define CC 1024
#define HH 16
#define DD 64

typedef __attribute__((ext_vector_type(16))) __bf16 v16bf;
typedef __attribute__((ext_vector_type(8)))  __bf16 v8bf;
typedef __attribute__((ext_vector_type(8)))  float  v8f;

union AFrag { v16bf v; v8bf h[2]; };

__device__ __forceinline__ v8f wmma_bf16(v16bf a, v16bf b, v8f c) {
  return __builtin_amdgcn_wmma_f32_16x16x32_bf16(
      /*neg_a=*/false, a, /*neg_b=*/false, b,
      /*c_mod=*/(short)0, c, /*reuse_a=*/false, /*reuse_b=*/false);
}

// A-matrix fragment (16x32 bf16), ISA layout: lane holds row M=lane&15,
// element j -> K = kbase + 16*(j>>3) + 8*half + (j&7)
// => two contiguous 16-byte chunks at kbase+8*half and kbase+16+8*half.
__device__ __forceinline__ v16bf load_a(const __bf16* base, int ld, long row,
                                        int kbase, int half) {
  const __bf16* p = base + (size_t)row * ld + kbase + 8 * half;
  AFrag u;
  u.h[0] = *(const v8bf*)(p);
  u.h[1] = *(const v8bf*)(p + 16);
  return u.v;
}

// B-matrix fragment (32x16 bf16) from memory laid out [n][k] with stride ld:
// lane holds column N=n, element j -> K = kbase + 16*(j>>3) + 8*half + (j&7).
__device__ __forceinline__ v16bf load_b(const __bf16* base, int ld, long n,
                                        int kbase, int half) {
  const __bf16* p = base + (size_t)n * ld + kbase + 8 * half;
  AFrag u;
  u.h[0] = *(const v8bf*)(p);
  u.h[1] = *(const v8bf*)(p + 16);
  return u.v;
}

__global__ void cvt_kernel(const float* __restrict__ s, __bf16* __restrict__ d,
                           int n) {
  int i = blockIdx.x * blockDim.x + threadIdx.x;
  int stride = gridDim.x * blockDim.x;
  for (; i < n; i += stride) d[i] = (__bf16)s[i];
}

// Transpose-convert: src f32 [K,N] row-major -> dst bf16 [N,K] row-major.
// 32x32 tiles through LDS; block = (32,8).
__global__ __launch_bounds__(256) void cvt_T_kernel(
    const float* __restrict__ s, __bf16* __restrict__ d, int K, int N) {
  __shared__ float tile[32][33];
  const int kt = blockIdx.y * 32, nt = blockIdx.x * 32;
  const int tx = threadIdx.x, ty = threadIdx.y;
  for (int i = 0; i < 32; i += 8)
    tile[ty + i][tx] = s[(size_t)(kt + ty + i) * N + nt + tx];
  __syncthreads();
  for (int i = 0; i < 32; i += 8)
    d[(size_t)(nt + ty + i) * K + kt + tx] = (__bf16)tile[tx][ty + i];
}

// C[M,N] = A[M,K] * BT^T ; A bf16 [M,K] row-major, BT bf16 [N,K] row-major
// (pre-transposed weights). No LDS, no barriers: both A and B fragments are
// K-contiguous in global memory; all tile re-reads hit the 192MB L2.
// Block: 128 threads = 4 waves; block tile 128x128; wave tile 64x64
// (4x4 WMMA accumulators -> 16 v_wmma per K=32 step).
__global__ __launch_bounds__(128) void gemm_bf16(
    const __bf16* __restrict__ A, const __bf16* __restrict__ BT, int K, int N,
    __bf16* __restrict__ Cb, float* __restrict__ Cf,
    const float* __restrict__ bias) {
  const int t = threadIdx.x;
  const int w = t >> 5, lane = t & 31, half = lane >> 4, l15 = lane & 15;
  const int wm = (w >> 1) * 64, wn = (w & 1) * 64;
  const long rowbase = (long)blockIdx.y * 128 + wm;
  const long colbase = (long)blockIdx.x * 128 + wn;

  v8f acc[4][4] = {};
  for (int k0 = 0; k0 < K; k0 += 32) {
    v16bf a[4];
#pragma unroll
    for (int mi = 0; mi < 4; mi++)
      a[mi] = load_a(A, K, rowbase + mi * 16 + l15, k0, half);
#pragma unroll
    for (int ni = 0; ni < 4; ni++) {
      v16bf b = load_b(BT, K, colbase + ni * 16 + l15, k0, half);
#pragma unroll
      for (int mi = 0; mi < 4; mi++)
        acc[mi][ni] = wmma_bf16(a[mi], b, acc[mi][ni]);
    }
  }
  // Epilogue. C layout: M = 8*half + r, N = lane&15.
#pragma unroll
  for (int mi = 0; mi < 4; mi++)
#pragma unroll
    for (int ni = 0; ni < 4; ni++) {
      long row = rowbase + mi * 16 + half * 8;
      long col = colbase + ni * 16 + l15;
      for (int r = 0; r < 8; r++) {
        float v = acc[mi][ni][r];
        size_t o = (size_t)(row + r) * N + col;
        if (Cf)
          Cf[o] = v + (bias ? bias[col] : 0.0f);
        else
          Cb[o] = (__bf16)v;
      }
    }
}

// Flash attention with permuted causal mask. One block = (b, h, 64 q rows),
// 4 waves x 16 q rows each. K tile staged in LDS as [n][d] via async
// global->LDS copies (ASYNCcnt path); V tile transposed [d][k] manually.
__global__ __launch_bounds__(128) void attn_kernel(
    const __bf16* __restrict__ qkv, const int* __restrict__ perm,
    __bf16* __restrict__ outb) {
  __shared__ __bf16 lk[64 * 64];      // lk[n*64 + d]
  __shared__ __bf16 lv[64 * 64];      // lv[d*64 + k]
  __shared__ __bf16 lp[4][16 * 64];   // per-wave probability tile [m][k]
  const int t = threadIdx.x;
  const int w = t >> 5, lane = t & 31, half = lane >> 4, l15 = lane & 15;

  const int blk = blockIdx.x;
  const int qt = blk & 31;          // T/64 = 32 query tiles
  const int h = (blk >> 5) & 15;
  const int b = blk >> 9;
  const int q0 = qt * 64;
  const size_t rowoff = (size_t)b * TT;
  const int ld = 3 * CC;
  const float scale = 0.125f;  // 1/sqrt(64)

  const __bf16* qbase = qkv + rowoff * ld + h * DD;
  const __bf16* kbase_g = qkv + rowoff * ld + CC + h * DD;
  const __bf16* vbase_g = qkv + rowoff * ld + 2 * CC + h * DD;

  // Resident q fragments: 16 rows x 64 K (two K=32 fragments).
  const long qrow = q0 + w * 16 + l15;
  v16bf qa0 = load_a(qbase, ld, qrow, 0, half);
  v16bf qa1 = load_a(qbase, ld, qrow, 32, half);

  float m_r[8], l_r[8];
  v8f o[4] = {};
  for (int r = 0; r < 8; r++) { m_r[r] = -3.0e38f; l_r[r] = 0.0f; }

  for (int kt = 0; kt < TT / 64; kt++) {
    const int kb = kt * 64;
    // Stage K tile with async global->LDS b128 copies; V tile transposed
    // manually (needs the register-level 8x8 scatter).
    for (int i8 = t; i8 < 64 * 64 / 8; i8 += 128) {
      int n = i8 >> 3;
      int d = (i8 & 7) * 8;
      unsigned lka = (unsigned)(size_t)&lk[n * 64 + d];  // LDS byte offset
      const __bf16* gp = &kbase_g[(size_t)(kb + n) * ld + d];
      asm volatile("global_load_async_to_lds_b128 %0, %1, off"
                   :: "v"(lka), "v"(gp) : "memory");
      v8bf vv = *(const v8bf*)&vbase_g[(size_t)(kb + n) * ld + d];
      for (int j = 0; j < 8; j++) lv[(d + j) * 64 + n] = vv[j];
    }
    asm volatile("s_wait_asynccnt 0x0" ::: "memory");
    __syncthreads();

    // scores s[16 q][64 k] = q . K^T  (B fragment contiguous over d in lk)
    v8f s[4];
    const int ibase = q0 + w * 16 + half * 8;
    for (int nt = 0; nt < 4; nt++) {
      v8f c = {};
      c = wmma_bf16(qa0, load_b(lk, 64, nt * 16 + l15, 0, half), c);
      c = wmma_bf16(qa1, load_b(lk, 64, nt * 16 + l15, 32, half), c);
      const int jg = kb + nt * 16 + l15;   // key column index (same for all r)
      const int pj = perm[jg];
      for (int r = 0; r < 8; r++) {
        float mb = (pj <= ibase + r) ? 0.0f : -1.0e9f;
        c[r] = c[r] * scale + mb;
      }
      s[nt] = c;
    }

    // Online softmax per row (row spans 16 lanes of one half + 4 n-tiles).
    for (int r = 0; r < 8; r++) {
      float pm = fmaxf(fmaxf(s[0][r], s[1][r]), fmaxf(s[2][r], s[3][r]));
      for (int off = 8; off; off >>= 1) pm = fmaxf(pm, __shfl_xor(pm, off, 32));
      float mn = fmaxf(m_r[r], pm);
      float corr = __expf(m_r[r] - mn);
      float rs = 0.0f;
      for (int nt = 0; nt < 4; nt++) {
        float p = __expf(s[nt][r] - mn);
        s[nt][r] = p;
        rs += p;
      }
      for (int off = 8; off; off >>= 1) rs += __shfl_xor(rs, off, 32);
      l_r[r] = l_r[r] * corr + rs;
      m_r[r] = mn;
      for (int dt = 0; dt < 4; dt++) o[dt][r] *= corr;
    }

    // C-layout -> A-layout transpose of p through per-wave LDS (as bf16).
    __bf16* lpw = lp[w];
    for (int nt = 0; nt < 4; nt++)
      for (int r = 0; r < 8; r++)
        lpw[(half * 8 + r) * 64 + nt * 16 + l15] = (__bf16)s[nt][r];
    __syncthreads();

    // o += p @ V  (B fragment contiguous over k in lv)
    for (int kk = 0; kk < 2; kk++) {
      v16bf pa = load_a(lpw, 64, l15, kk * 32, half);
      for (int dt = 0; dt < 4; dt++)
        o[dt] = wmma_bf16(pa, load_b(lv, 64, dt * 16 + l15, kk * 32, half), o[dt]);
    }
    __syncthreads();
  }

  // Normalize and write attn output (bf16, [B*T, C] with head-strided cols).
  const long orow = (long)rowoff + q0 + w * 16 + half * 8;
  float inv[8];
  for (int r = 0; r < 8; r++) inv[r] = 1.0f / l_r[r];
  for (int dt = 0; dt < 4; dt++)
    for (int r = 0; r < 8; r++)
      outb[(size_t)(orow + r) * CC + h * DD + dt * 16 + l15] =
          (__bf16)(o[dt][r] * inv[r]);
}

extern "C" void kernel_launch(void* const* d_in, const int* in_sizes, int n_in,
                              void* d_out, int out_size, void* d_ws,
                              size_t ws_size, hipStream_t stream) {
  (void)in_sizes; (void)n_in; (void)out_size; (void)ws_size;
  const float* x    = (const float*)d_in[0];
  const float* Wqkv = (const float*)d_in[1];
  const float* Wout = (const float*)d_in[2];
  const float* bout = (const float*)d_in[3];
  const int*   perm = (const int*)d_in[4];
  float* out = (float*)d_out;

  char* ws = (char*)d_ws;
  __bf16* xb    = (__bf16*)(ws);                          // 16 MB  [B*T, C]
  __bf16* wqbT  = (__bf16*)(ws + (size_t)(16u << 20));    //  6 MB  [3C, C]
  __bf16* wobT  = (__bf16*)(ws + (size_t)(22u << 20));    //  2 MB  [C, C]
  __bf16* qkvb  = (__bf16*)(ws + (size_t)(24u << 20));    // 48 MB  [B*T, 3C]
  __bf16* attnb = (__bf16*)(ws + (size_t)(72u << 20));    // 16 MB  [B*T, C]

  cvt_kernel<<<2048, 256, 0, stream>>>(x, xb, BB * TT * CC);
  {
    dim3 bt(32, 8);
    dim3 gq(3 * CC / 32, CC / 32);   // Wqkv: [C, 3C] -> [3C, C]
    cvt_T_kernel<<<gq, bt, 0, stream>>>(Wqkv, wqbT, CC, 3 * CC);
    dim3 go(CC / 32, CC / 32);       // Wout: [C, C] -> [C, C] transposed
    cvt_T_kernel<<<go, bt, 0, stream>>>(Wout, wobT, CC, CC);
  }

  dim3 g1(3 * CC / 128, BB * TT / 128);
  gemm_bf16<<<g1, 128, 0, stream>>>(xb, wqbT, CC, 3 * CC, qkvb, nullptr,
                                    nullptr);

  attn_kernel<<<BB * HH * (TT / 64), 128, 0, stream>>>(qkvb, perm, attnb);

  dim3 g2(CC / 128, BB * TT / 128);
  gemm_bf16<<<g2, 128, 0, stream>>>(attnb, wobT, CC, CC, nullptr, out, bout);
}
